// AttentionSinCos_Separate_jointMLP_26096221291135
// MI455X (gfx1250) — matrix operand order repaired and checked by reading, not compiled, not run on hardware
//
#include <hip/hip_runtime.h>
#include <hip/hip_bf16.h>
#include <math.h>

// ---------------------------------------------------------------------------
// CDNA5 (gfx1250) implementation: all GEMMs via v_wmma_f32_16x16x32_f16,
// attention fused flash-style (scores never hit HBM), X-tile staging through
// the gfx1250 async global->LDS DMA path.
// ---------------------------------------------------------------------------

typedef __attribute__((ext_vector_type(16))) _Float16 v16h;
typedef __attribute__((ext_vector_type(8)))  float    v8f;
typedef __attribute__((ext_vector_type(4)))  int      v4i;

#ifndef ASYNC_LDS
#if defined(__HIP_DEVICE_COMPILE__) && __has_builtin(__builtin_amdgcn_global_load_async_to_lds_b128)
#define ASYNC_LDS 1
#else
#define ASYNC_LDS 0
#endif
#endif

// Load a v16h fragment as two 16B pieces (matches A/B VGPR layouts; all
// addresses are 16B aligned by construction).
__device__ __forceinline__ v16h ld16(const _Float16* lo, const _Float16* hi) {
    union { uint4 u[2]; v16h v; } t;
    t.u[0] = *(const uint4*)lo;
    t.u[1] = *(const uint4*)hi;
    return t.v;
}

__device__ __forceinline__ v8f wmma_f16(v16h a, v16h b, v8f c) {
    return __builtin_amdgcn_wmma_f32_16x16x32_f16(false, a, false, b,
                                                  (short)0, c, false, false);
}

// reductions across a 16-lane half of the wave (xor masks 1..8 never cross bit4)
__device__ __forceinline__ float hmax16(float v) {
    for (int o = 8; o; o >>= 1) v = fmaxf(v, __shfl_xor(v, o, 32));
    return v;
}
__device__ __forceinline__ float hsum16(float v) {
    for (int o = 8; o; o >>= 1) v += __shfl_xor(v, o, 32);
    return v;
}

// ---------------------------------------------------------------------------
// f32 -> f16 convert with zero-padding of the K dimension to Kp (mult. of 64)
// ---------------------------------------------------------------------------
__global__ void cvt_pad_kernel(const float* __restrict__ src, _Float16* __restrict__ dst,
                               int Kd, int Kp, long total) {
    long i = (long)blockIdx.x * blockDim.x + threadIdx.x;
    if (i >= total) return;
    long r = i / Kp;
    int  c = (int)(i - r * Kp);
    dst[i] = (c < Kd) ? (_Float16)src[r * Kd + c] : (_Float16)0.f;
}

// ---------------------------------------------------------------------------
// WMMA GEMM: out[M,Nd] = X[M,Kp] @ W[Nd,Kp]^T (+bias) (+silu) (+residual)
// Block = 256 threads = 8 waves; tile 32(M) x 64(N); K streamed in 64-chunks.
// ---------------------------------------------------------------------------
#define ACT_NONE 0
#define ACT_SILU 1

__global__ __launch_bounds__(256) void gemm_wmma_kernel(
        const _Float16* __restrict__ X, const _Float16* __restrict__ W,
        const float* __restrict__ bias, const float* __restrict__ residual,
        float* __restrict__ out, int Nd, int Kp, int act) {
    __shared__ __align__(16) _Float16 Xs[32 * 64];
    __shared__ __align__(16) _Float16 Ws[64 * 64];

    const int tid  = threadIdx.x;
    const int wid  = tid >> 5;
    const int lane = tid & 31;
    const int mi   = wid >> 2;          // 0..1 : 16-row subtile
    const int ni   = wid & 3;           // 0..3 : 16-col subtile
    const int col  = lane & 15;
    const int half = (lane >> 4) & 1;
    const long mbase = (long)blockIdx.x * 32;
    const int  nbase = blockIdx.y * 64;

    v8f acc = {};
    for (int kc = 0; kc < Kp; kc += 64) {
        {   // stage X tile: 32 rows x 64 halves, one b128 per thread
            int r  = tid >> 3;
            int c8 = (tid & 7) * 8;
            const _Float16* src = X + (size_t)(mbase + r) * Kp + kc + c8;
#if ASYNC_LDS
            // gfx1250 async DMA: (as1 v4i* gsrc, as3 v4i* lds, imm offset, imm cpol)
            __builtin_amdgcn_global_load_async_to_lds_b128(
                (__attribute__((address_space(1))) v4i*)src,
                (__attribute__((address_space(3))) v4i*)&Xs[r * 64 + c8], 0, 0);
#else
            *(uint4*)&Xs[r * 64 + c8] = *(const uint4*)src;
#endif
        }
        {   // stage W tile: 64 out-cols x 64 halves, two b128 per thread
            int r    = tid >> 2;
            int c8   = (tid & 3) * 16;
            int ncol = nbase + r;
            if (ncol < Nd) {
                const _Float16* src = W + (size_t)ncol * Kp + kc + c8;
                *(uint4*)&Ws[r * 64 + c8]     = *(const uint4*)(src);
                *(uint4*)&Ws[r * 64 + c8 + 8] = *(const uint4*)(src + 8);
            } else {
                uint4 zz = {0u, 0u, 0u, 0u};
                *(uint4*)&Ws[r * 64 + c8]     = zz;
                *(uint4*)&Ws[r * 64 + c8 + 8] = zz;
            }
        }
#if ASYNC_LDS
#if __has_builtin(__builtin_amdgcn_s_wait_asynccnt)
        __builtin_amdgcn_s_wait_asynccnt(0);
#else
        asm volatile("s_wait_asynccnt 0" ::: "memory");
#endif
#endif
        __syncthreads();
        #pragma unroll
        for (int kk = 0; kk < 64; kk += 32) {
            // A frag: lane<16 -> K kk..kk+15 of row col; lane>=16 -> kk+16..kk+31
            v16h a = ld16(&Xs[(mi * 16 + col) * 64 + kk + half * 16],
                          &Xs[(mi * 16 + col) * 64 + kk + half * 16 + 8]);
            // B frag: lane<16 -> K kk..kk+15 of column col; lane>=16 -> kk+16..kk+31
            v16h b = ld16(&Ws[(ni * 16 + col) * 64 + kk + half * 16],
                          &Ws[(ni * 16 + col) * 64 + kk + half * 16 + 8]);
            acc = wmma_f16(a, b, acc);
        }
        __syncthreads();
    }
    // C layout: VGPR g -> row g + 8*half, lane%16 -> column
    int ncol = nbase + ni * 16 + col;
    if (ncol < Nd) {
        float bval = bias ? bias[ncol] : 0.f;
        #pragma unroll
        for (int g = 0; g < 8; ++g) {
            long r = mbase + mi * 16 + g + half * 8;
            float v = acc[g] + bval;
            if (act == ACT_SILU) v = v / (1.f + __expf(-v));
            if (residual) v += residual[r * Nd + ncol];
            out[r * Nd + ncol] = v;
        }
    }
}

// ---------------------------------------------------------------------------
// Fused flash attention. One wave per 16-query tile of one (batch, head).
// hd in {8,10,20}, zero-padded to 32. Scores stay in VGPRs/LDS.
// ---------------------------------------------------------------------------
__global__ __launch_bounds__(256) void attn_kernel(
        const float* __restrict__ Q, const float* __restrict__ Kk,
        const float* __restrict__ V, const unsigned char* __restrict__ mask,
        float* __restrict__ out, int Nn, int E, int H, int hd) {
    __shared__ __align__(16) _Float16 sm[8 * 3072];   // 6KB per wave
    const int tid  = threadIdx.x;
    const int wid  = tid >> 5;
    const int lane = tid & 31;
    _Float16* Qs = sm + wid * 3072;     // 16x32
    _Float16* Ks = Qs + 512;            // 32 keys x 32 dims (row-major)
    _Float16* Vt = Ks + 1024;           // 32 dims  x 32 keys (transposed)
    _Float16* Ps = Vt + 1024;           // 16x32 probabilities

    const int tilesPerBH = Nn >> 4;
    int wg = blockIdx.x * 8 + wid;
    int qt = wg % tilesPerBH;
    int bh = wg / tilesPerBH;
    int h  = bh % H;
    int b  = bh / H;

    const float scale = rsqrtf((float)hd);
    const int col  = lane & 15;
    const int half = (lane >> 4) & 1;

    {   // stage Q tile (scaled, f16, zero-padded dims)
        int r  = lane >> 1;
        int d0 = (lane & 1) * 16;
        const float* qrow = Q + ((size_t)(b * Nn) + qt * 16 + r) * E + h * hd;
        for (int j = 0; j < 16; ++j) {
            int d = d0 + j;
            float v = (d < hd) ? qrow[d] * scale : 0.f;
            Qs[r * 32 + d] = (_Float16)v;
        }
    }
    v16h aq = ld16(&Qs[col * 32 + half * 16], &Qs[col * 32 + half * 16 + 8]);

    float mrow[8], lrow[8];
    #pragma unroll
    for (int g = 0; g < 8; ++g) { mrow[g] = -1e30f; lrow[g] = 0.f; }
    v8f acc0 = {}; v8f acc1 = {};

    const int nkb = Nn >> 5;
    for (int kb = 0; kb < nkb; ++kb) {
        int k0 = kb * 32;
        {   // stage K rows + V transposed (one key per lane)
            int key = k0 + lane;
            const float* krow = Kk + ((size_t)(b * Nn) + key) * E + h * hd;
            const float* vrow = V  + ((size_t)(b * Nn) + key) * E + h * hd;
            for (int d = 0; d < 32; ++d) {
                float kv = (d < hd) ? krow[d] : 0.f;
                float vv = (d < hd) ? vrow[d] : 0.f;
                Ks[lane * 32 + d] = (_Float16)kv;
                Vt[d * 32 + lane] = (_Float16)vv;
            }
        }
        int mk  = (int)mask[(size_t)b * Nn + k0 + lane];
        int m0  = __shfl(mk, col, 32);        // mask for column keys k0+col
        int m1v = __shfl(mk, 16 + col, 32);   // and k0+16+col

        v16h bk0 = ld16(&Ks[col * 32 + half * 16],        &Ks[col * 32 + half * 16 + 8]);
        v16h bk1 = ld16(&Ks[(16 + col) * 32 + half * 16], &Ks[(16 + col) * 32 + half * 16 + 8]);
        v8f z = {};
        v8f S0 = wmma_f16(aq, bk0, z);
        v8f S1 = wmma_f16(aq, bk1, z);

        #pragma unroll
        for (int g = 0; g < 8; ++g) {
            float s0 = m0  ? -1e30f : S0[g];
            float s1 = m1v ? -1e30f : S1[g];
            float mx = hmax16(fmaxf(s0, s1));
            float mnew  = fmaxf(mrow[g], mx);
            float alpha = __expf(mrow[g] - mnew);
            float p0 = __expf(s0 - mnew);
            float p1 = __expf(s1 - mnew);
            float ps = hsum16(p0 + p1);
            lrow[g] = lrow[g] * alpha + ps;
            mrow[g] = mnew;
            acc0[g] *= alpha;
            acc1[g] *= alpha;
            // transpose P through LDS: row = g+8*half, column = key
            Ps[(g + half * 8) * 32 + col]      = (_Float16)p0;
            Ps[(g + half * 8) * 32 + 16 + col] = (_Float16)p1;
        }
        v16h ap  = ld16(&Ps[col * 32 + half * 16], &Ps[col * 32 + half * 16 + 8]);
        v16h bv0 = ld16(&Vt[col * 32 + half * 16], &Vt[col * 32 + half * 16 + 8]);
        acc0 = wmma_f16(ap, bv0, acc0);
        if (hd > 16) {   // wave-uniform
            v16h bv1 = ld16(&Vt[(16 + col) * 32 + half * 16],
                            &Vt[(16 + col) * 32 + half * 16 + 8]);
            acc1 = wmma_f16(ap, bv1, acc1);
        }
    }
    #pragma unroll
    for (int g = 0; g < 8; ++g) {
        int r = qt * 16 + g + half * 8;
        float inv = 1.f / lrow[g];
        float* orow = out + ((size_t)(b * Nn) + r) * E + h * hd;
        if (col < hd)      orow[col]      = acc0[g] * inv;
        if (16 + col < hd) orow[16 + col] = acc1[g] * inv;
    }
}

// ---------------------------------------------------------------------------
// Elementwise / per-row kernels
// ---------------------------------------------------------------------------
__global__ void ln_kernel(const float* __restrict__ x, const float* __restrict__ g,
                          const float* __restrict__ bta, float* __restrict__ o,
                          int D, long Mrows) {
    long m = (long)blockIdx.x * blockDim.x + threadIdx.x;
    if (m >= Mrows) return;
    const float* xr = x + m * D;
    float* orow = o + m * D;
    float mu = 0.f;
    for (int d = 0; d < D; ++d) mu += xr[d];
    mu /= D;
    float var = 0.f;
    for (int d = 0; d < D; ++d) { float t = xr[d] - mu; var += t * t; }
    var /= D;
    float inv = rsqrtf(var + 1e-5f);
    for (int d = 0; d < D; ++d) orow[d] = (xr[d] - mu) * inv * g[d] + bta[d];
}

// builds q_in[M,80] = [atoms_e | pf] and kv_in[M,112] = [atoms_e | pf | lat_e]
// pf[2r]/[2r+1] = sin/cos(2*pi*(i+1)*x_j), r = 3i+j  (Wsc_q==Wsc_k==Wsc_v)
__global__ void build_inputs_kernel(const float* __restrict__ frac,
                                    const float* __restrict__ atoms_e,
                                    const float* __restrict__ lat_e,
                                    float* __restrict__ q_in, float* __restrict__ kv_in,
                                    long Mrows) {
    long m = (long)blockIdx.x * blockDim.x + threadIdx.x;
    if (m >= Mrows) return;
    float fx[3];
    fx[0] = frac[m * 3 + 0]; fx[1] = frac[m * 3 + 1]; fx[2] = frac[m * 3 + 2];
    float* q  = q_in  + m * 80;
    float* kv = kv_in + m * 112;
    for (int d = 0; d < 32; ++d) { float a = atoms_e[m * 32 + d]; q[d] = a; kv[d] = a; }
    for (int i = 0; i < 8; ++i)
        for (int j = 0; j < 3; ++j) {
            int r = 3 * i + j;
            float ang = 6.28318530717958647692f * (float)(i + 1) * fx[j];
            float s, c;
            __sincosf(ang, &s, &c);
            q[32 + 2 * r] = s;  q[33 + 2 * r] = c;
            kv[32 + 2 * r] = s; kv[33 + 2 * r] = c;
        }
    for (int d = 0; d < 32; ++d) kv[80 + d] = lat_e[m * 32 + d];
}

__global__ void concat_kv_kernel(const float* __restrict__ n_norm,
                                 const float* __restrict__ l_norm,
                                 float* __restrict__ kv, long Mrows) {
    long m = (long)blockIdx.x * blockDim.x + threadIdx.x;
    if (m >= Mrows) return;
    float* o = kv + m * 112;
    for (int d = 0; d < 80; ++d) o[d] = n_norm[m * 80 + d];
    for (int d = 0; d < 32; ++d) o[80 + d] = l_norm[m * 32 + d];
}

__global__ void copy_kernel(const float* __restrict__ s, float* __restrict__ d, long n) {
    long i = (long)blockIdx.x * blockDim.x + threadIdx.x;
    if (i < n) d[i] = s[i];
}

// ---------------------------------------------------------------------------
// Host orchestration
// ---------------------------------------------------------------------------
extern "C" void kernel_launch(void* const* d_in, const int* in_sizes, int n_in,
                              void* d_out, int out_size, void* d_ws, size_t ws_size,
                              hipStream_t stream) {
    (void)in_sizes; (void)n_in; (void)out_size; (void)ws_size;
    const int  Bb = 64, Nn = 512, H = 4;
    const long M = (long)Bb * Nn;

    const float* frac = (const float*)d_in[0];
    const float* atom = (const float*)d_in[1];
    const float* latt = (const float*)d_in[2];
    const unsigned char* mask = (const unsigned char*)d_in[3];
    auto F = [&](int i) { return (const float*)d_in[i]; };
    // param leaf order (jax pytree insertion order):
    // 4 emb_a, 5 emb_l, 6/7 atom_ln g/b, 8/9 lat_ln g/b, 10-12 Wsc_{q,k,v},
    // 13/14 Wq_out w/b, 15/16 Wk_out w/b, 17/18 Wv_out w/b,
    // 19..26 sc_attn {qw,kw,vw,qb,kb,vb,ow,ob},
    // 27..114 node_layers[4] x 22 leaves, 115..202 lat_layers[4] x 22 leaves.
    // per-layer: +0/1 Wq, +2/3 Wk, +4/5 Wv, +6 qw,+7 kw,+8 vw,+9 qb,+10 kb,
    // +11 vb,+12 ow,+13 ob, +14/15 ln, +16/17 pre_ln, +18/19 m1, +20/21 m2

    char* cur = (char*)d_ws;
    auto aballoc = [&](long bytes) { char* p = cur; cur += (bytes + 255) & ~255L; return p; };
    auto allocf  = [&](long n) { return (float*)aballoc(n * 4); };
    auto alloch  = [&](long n) { return (_Float16*)aballoc(n * 2); };

    _Float16* xh = alloch(M * 320);
    _Float16* wh = alloch(320 * 384);

    auto gemm = [&](const float* X, int Kd, const float* W, const float* bias,
                    int Nd, float* out, int act, const float* res) {
        int Kp = (Kd + 63) & ~63;
        long tx = M * Kp;
        cvt_pad_kernel<<<dim3((unsigned)((tx + 255) / 256)), dim3(256), 0, stream>>>(X, xh, Kd, Kp, tx);
        long tw = (long)Nd * Kp;
        cvt_pad_kernel<<<dim3((unsigned)((tw + 255) / 256)), dim3(256), 0, stream>>>(W, wh, Kd, Kp, tw);
        dim3 g((unsigned)(M / 32), (unsigned)((Nd + 63) / 64));
        gemm_wmma_kernel<<<g, dim3(256), 0, stream>>>(xh, wh, bias, res, out, Nd, Kp, act);
    };
    auto lnf = [&](const float* x, const float* g, const float* b, int D, float* o) {
        ln_kernel<<<dim3((unsigned)(M / 256)), dim3(256), 0, stream>>>(x, g, b, o, D, M);
    };
    auto attention = [&](const float* q, const float* k, const float* v,
                         float* o, int E, int Hh) {
        int hd = E / Hh;
        int blocks = Bb * Hh * (Nn / 16) / 8;
        attn_kernel<<<dim3(blocks), dim3(256), 0, stream>>>(q, k, v, mask, o, Nn, E, Hh, hd);
    };

    // workspace buffers
    float* atoms_pre = allocf(M * 32);
    float* atoms_e   = allocf(M * 32);
    float* lat_pre   = allocf(M * 32);
    float* q_in      = allocf(M * 80);
    float* kv_in     = allocf(M * 112);
    float* qS        = allocf(M * 80);
    float* kS        = allocf(M * 112);
    float* vS        = allocf(M * 112);
    float* qhB       = allocf(M * 80);
    float* khB       = allocf(M * 80);
    float* vhB       = allocf(M * 80);
    float* attnB     = allocf(M * 80);
    float* nodes0    = allocf(M * 80);
    float* nodes1    = allocf(M * 80);
    float* lat0      = allocf(M * 32);
    float* lat1      = allocf(M * 32);
    float* n_norm    = allocf(M * 80);
    float* l_norm    = allocf(M * 32);
    float* kvbuf     = allocf(M * 112);
    float* q1        = allocf(M * 80);
    float* k1        = allocf(M * 112);
    float* v1        = allocf(M * 112);
    float* xB        = allocf(M * 80);
    float* rB        = allocf(M * 80);
    float* hidB      = allocf(M * 320);

    // ---- embeddings ----
    gemm(atom, 128, F(4), nullptr, 32, atoms_pre, ACT_NONE, nullptr);
    lnf(atoms_pre, F(6), F(7), 32, atoms_e);
    gemm(latt, 6, F(5), nullptr, 32, lat_pre, ACT_NONE, nullptr);
    lnf(lat_pre, F(8), F(9), 32, lat0);                    // lat0 = lat_e

    // ---- periodic encoding + concatenated q/kv inputs (qf==kf==vf) ----
    build_inputs_kernel<<<dim3((unsigned)(M / 256)), dim3(256), 0, stream>>>(
        frac, atoms_e, lat0, q_in, kv_in, M);

    // ---- sin/cos attention block ----
    gemm(q_in, 80,  F(13), F(14), 80,  qS, ACT_NONE, nullptr);   // Wq_out
    gemm(kv_in, 112, F(15), F(16), 112, kS, ACT_NONE, nullptr);  // Wk_out
    gemm(kv_in, 112, F(17), F(18), 112, vS, ACT_NONE, nullptr);  // Wv_out
    gemm(qS, 80,  F(19), F(22), 80, qhB, ACT_NONE, nullptr);     // sc qw
    gemm(kS, 112, F(20), F(23), 80, khB, ACT_NONE, nullptr);     // sc kw
    gemm(vS, 112, F(21), F(24), 80, vhB, ACT_NONE, nullptr);     // sc vw
    attention(qhB, khB, vhB, attnB, 80, 8);                      // K=8 heads, hd=10
    gemm(attnB, 80, F(25), F(26), 80, nodes0, ACT_NONE, nullptr);// sc ow

    float* nodesCur = nodes0; float* nodesNext = nodes1;
    float* latCur = lat0;     float* latNext = lat1;

    auto run_layer = [&](int base, int in_d, int hidden, const float* x_norm,
                         float* outB, int Hh) {
        gemm(x_norm, in_d, F(base + 0), F(base + 1), in_d, q1, ACT_NONE, nullptr);
        gemm(kvbuf, 112, F(base + 2), F(base + 3), 112, k1, ACT_NONE, nullptr);
        gemm(kvbuf, 112, F(base + 4), F(base + 5), 112, v1, ACT_NONE, nullptr);
        gemm(q1, in_d, F(base + 6), F(base + 9),  in_d, qhB, ACT_NONE, nullptr);
        gemm(k1, 112,  F(base + 7), F(base + 10), in_d, khB, ACT_NONE, nullptr);
        gemm(v1, 112,  F(base + 8), F(base + 11), in_d, vhB, ACT_NONE, nullptr);
        attention(qhB, khB, vhB, attnB, in_d, Hh);
        gemm(attnB, in_d, F(base + 12), F(base + 13), in_d, xB, ACT_NONE, q1); // x = q1 + proj
        lnf(xB, F(base + 14), F(base + 15), in_d, rB);
        gemm(rB, in_d, F(base + 18), F(base + 19), hidden, hidB, ACT_SILU, nullptr);
        gemm(hidB, hidden, F(base + 20), F(base + 21), in_d, outB, ACT_NONE, xB); // x + r
    };

    for (int i = 0; i < 4; ++i) {
        int nb = 27 + 22 * i;
        int lb = 115 + 22 * i;
        lnf(nodesCur, F(nb + 16), F(nb + 17), 80, n_norm);
        lnf(latCur,   F(lb + 16), F(lb + 17), 32, l_norm);
        concat_kv_kernel<<<dim3((unsigned)(M / 256)), dim3(256), 0, stream>>>(
            n_norm, l_norm, kvbuf, M);
        run_layer(nb, 80, 320, n_norm, nodesNext, H);
        run_layer(lb, 32, 128, l_norm, latNext, H);
        float* t;
        t = nodesCur; nodesCur = nodesNext; nodesNext = t;
        t = latCur;   latCur = latNext;     latNext = t;
    }

    // ---- outputs: (nodes, lat) concatenated flat ----
    float* outp = (float*)d_out;
    long n0 = M * 80, n1 = M * 32;
    copy_kernel<<<dim3((unsigned)((n0 + 255) / 256)), dim3(256), 0, stream>>>(nodesCur, outp, n0);
    copy_kernel<<<dim3((unsigned)((n1 + 255) / 256)), dim3(256), 0, stream>>>(latCur, outp + n0, n1);
}
